// RNNDecoder_4011499454504
// MI455X (gfx1250) — compile-verified, matrix operand order checked by reading, not run on hardware
//
#include <hip/hip_runtime.h>
#include <hip/hip_bf16.h>

#define B_     128
#define TS_    512
#define TC_    128
#define ENC_   128
#define H_     1024
#define G_     4096
#define KA_    10
#define NOUT_  121

#define KT0    37   // ceil((3+128+1024)/32)  = ceil(1155/32)
#define KTN    69   // ceil((3+128+1024+1024)/32) = ceil(2179/32)
#define KTO    96   // 3072/32

typedef __attribute__((ext_vector_type(16))) __bf16 v16bf;
typedef __attribute__((ext_vector_type(8)))  float  v8f;

// ---------- helpers ----------
__device__ __forceinline__ unsigned short f2bf(float f) {
  union { float f; unsigned u; } v; v.f = f;
  unsigned r = v.u + 0x7FFFu + ((v.u >> 16) & 1u);   // round-to-nearest-even
  return (unsigned short)(r >> 16);
}
// CDNA5 16-bit A/B fragment K mapping: lane L (grp=L>>4), element e in 0..15
__device__ __forceinline__ int kmap(int e, int grp) {
  return (e & 7) + ((e >> 3) << 4) + grp * 8;
}
__device__ __forceinline__ float sigf(float x) { return 1.0f / (1.0f + __expf(-x)); }

// ---------- weight / input packing ----------
__global__ __launch_bounds__(256) void pack_w0(const float* __restrict__ Wih,
                                               const float* __restrict__ Whh,
                                               unsigned short* __restrict__ Wp) {
  int id = blockIdx.x * blockDim.x + threadIdx.x;   // 256*KT0*512 exact
  int e = id & 15, lane = (id >> 4) & 31, rest = id >> 9;
  int kt = rest % KT0, gt = rest / KT0;
  int g = gt * 16 + (lane & 15);
  int k = kt * 32 + kmap(e, lane >> 4);
  float v = 0.0f;
  if (k < 131)        v = Wih[g * 131 + k];
  else if (k < 1155)  v = Whh[g * 1024 + (k - 131)];
  Wp[id] = f2bf(v);
}

__global__ __launch_bounds__(256) void pack_wn(const float* __restrict__ Wih,
                                               const float* __restrict__ Whh,
                                               unsigned short* __restrict__ Wp) {
  int id = blockIdx.x * blockDim.x + threadIdx.x;   // 256*KTN*512 exact
  int e = id & 15, lane = (id >> 4) & 31, rest = id >> 9;
  int kt = rest % KTN, gt = rest / KTN;
  int g = gt * 16 + (lane & 15);
  int k = kt * 32 + kmap(e, lane >> 4);
  float v = 0.0f;
  if (k < 1155)       v = Wih[(size_t)g * 1155 + k];
  else if (k < 2179)  v = Whh[g * 1024 + (k - 1155)];
  Wp[id] = f2bf(v);
}

__global__ __launch_bounds__(256) void pack_wout(const float* __restrict__ Wout,
                                                 unsigned short* __restrict__ Wp) {
  int id = blockIdx.x * blockDim.x + threadIdx.x;   // 8*KTO*512 exact
  int e = id & 15, lane = (id >> 4) & 31, rest = id >> 9;
  int kt = rest % KTO, nt = rest / KTO;
  int n = nt * 16 + (lane & 15);
  int k = kt * 32 + kmap(e, lane >> 4);
  float v = (n < NOUT_) ? Wout[(size_t)n * 3072 + k] : 0.0f;
  Wp[id] = f2bf(v);
}

__global__ __launch_bounds__(256) void prep_bias(const float* __restrict__ bi0,
                                                 const float* __restrict__ bh0,
                                                 const float* __restrict__ bin,
                                                 const float* __restrict__ bhn,
                                                 float* __restrict__ bias0,
                                                 float* __restrict__ biasn) {
  int id = blockIdx.x * blockDim.x + threadIdx.x;   // 3*G_ exact
  if (id < G_) bias0[id] = bi0[id] + bh0[id];
  else { int j = id - G_; biasn[j] = bin[j] + bhn[j]; }
}

__global__ __launch_bounds__(256) void prep_ctx(const float* __restrict__ ctx,
                                                const float* __restrict__ mask,
                                                float* __restrict__ ctxm) {
  int id = blockIdx.x * blockDim.x + threadIdx.x;   // B*TC*ENC exact
  ctxm[id] = ctx[id] * mask[id >> 7];
}

__global__ __launch_bounds__(256) void init_state(const float* __restrict__ h0,
                                                  const float* __restrict__ c0,
                                                  const float* __restrict__ w0,
                                                  const float* __restrict__ k0,
                                                  float* __restrict__ hs, float* __restrict__ cs,
                                                  float* __restrict__ wst, float* __restrict__ kst) {
  int id = blockIdx.x * blockDim.x + threadIdx.x;
  const int NH = 3 * B_ * H_;
  if (id < NH) {
    hs[id] = h0[(id / (B_ * H_)) * H_ + (id % H_)];
  } else if (id < 2 * NH) {
    int i2 = id - NH;
    cs[i2] = c0[(i2 / (B_ * H_)) * H_ + (i2 % H_)];
  } else if (id < 2 * NH + B_ * ENC_) {
    int i2 = id - 2 * NH;
    wst[i2] = w0[i2 & (ENC_ - 1)];
  } else if (id < 2 * NH + B_ * ENC_ + B_ * KA_) {
    int i2 = id - 2 * NH - B_ * ENC_;
    kst[i2] = k0[i2 % KA_];
  }
}

// ---------- per-step A-fragment assembly (gather -> bf16 WMMA layout) ----------
__global__ __launch_bounds__(256) void assemble0(const float* __restrict__ strokes,
                                                 const float* __restrict__ wst,
                                                 const float* __restrict__ h0,
                                                 unsigned short* __restrict__ Ap, int t) {
  int id = blockIdx.x * blockDim.x + threadIdx.x;   // 8*KT0*512 exact
  int e = id & 15, lane = (id >> 4) & 31, rest = id >> 9;
  int kt = rest % KT0, mt = rest / KT0;
  int row = mt * 16 + (lane & 15);
  int k = kt * 32 + kmap(e, lane >> 4);
  float v = 0.0f;
  if (k < 3)          v = strokes[(row * TS_ + t) * 3 + k];
  else if (k < 131)   v = wst[row * ENC_ + (k - 3)];
  else if (k < 1155)  v = h0[row * H_ + (k - 131)];
  Ap[id] = f2bf(v);
}

__global__ __launch_bounds__(256) void assemble12(const float* __restrict__ strokes,
                                                  const float* __restrict__ wst,
                                                  const float* __restrict__ hA,
                                                  const float* __restrict__ hB,
                                                  unsigned short* __restrict__ Ap, int t) {
  int id = blockIdx.x * blockDim.x + threadIdx.x;   // 8*KTN*512 exact
  int e = id & 15, lane = (id >> 4) & 31, rest = id >> 9;
  int kt = rest % KTN, mt = rest / KTN;
  int row = mt * 16 + (lane & 15);
  int k = kt * 32 + kmap(e, lane >> 4);
  float v = 0.0f;
  if (k < 3)          v = strokes[(row * TS_ + t) * 3 + k];
  else if (k < 131)   v = wst[row * ENC_ + (k - 3)];
  else if (k < 1155)  v = hA[row * H_ + (k - 131)];
  else if (k < 2179)  v = hB[row * H_ + (k - 1155)];
  Ap[id] = f2bf(v);
}

__global__ __launch_bounds__(256) void assembleOut(const float* __restrict__ hs,
                                                   unsigned short* __restrict__ Ap) {
  int id = blockIdx.x * blockDim.x + threadIdx.x;   // 8*KTO*512 exact
  int e = id & 15, lane = (id >> 4) & 31, rest = id >> 9;
  int kt = rest % KTO, mt = rest / KTO;
  int row = mt * 16 + (lane & 15);
  int k = kt * 32 + kmap(e, lane >> 4);
  Ap[id] = f2bf(hs[(size_t)(k >> 10) * (B_ * H_) + row * H_ + (k & 1023)]);
}

// ---------- fused WMMA GEMM + LSTM cell ----------
// One wave owns a 16(batch) x 16(hidden) tile for ALL FOUR gates; the cell
// nonlinearity runs in registers right after the K loop.  KT is a template
// parameter so the inter-gate stride is a compile-time 24-bit immediate and
// the inner loop needs only three pointer bumps.
template <int KT>
__global__ __launch_bounds__(256) void lstm_gemm(const unsigned short* __restrict__ Apack,
                                                 const unsigned short* __restrict__ Wpack,
                                                 const float* __restrict__ bias,
                                                 float* __restrict__ hst,
                                                 float* __restrict__ cst) {
  const int wave  = threadIdx.x >> 5;
  const int lane  = threadIdx.x & 31;
  const int mt    = blockIdx.x >> 3;               // 0..7 batch tiles
  const int htile = (blockIdx.x & 7) * 8 + wave;   // 0..63 hidden tiles
  constexpr int GS = 64 * KT * 1024;               // bytes between gate blocks (< 2^23)
  v8f acc0 = {}, acc1 = {}, acc2 = {}, acc3 = {};
  const char* ap  = (const char*)Apack + ((size_t)(mt * KT) * 32 + lane) * 32;
  const char* bp0 = (const char*)Wpack + (((size_t)htile * KT) * 32 + lane) * 32;
  const char* bp2 = bp0 + 2 * (size_t)GS;
#pragma unroll 2
  for (int kt = 0; kt < KT; ++kt) {
    v16bf a  = *(const v16bf*)ap;
    v16bf b0 = *(const v16bf*)(bp0);
    v16bf b1 = *(const v16bf*)(bp0 + GS);
    v16bf b2 = *(const v16bf*)(bp2);
    v16bf b3 = *(const v16bf*)(bp2 + GS);
    __builtin_prefetch((const void*)(ap + 1024), 0, 1);
    acc0 = __builtin_amdgcn_wmma_f32_16x16x32_bf16(false, a, false, b0, (short)0, acc0, false, false);
    acc1 = __builtin_amdgcn_wmma_f32_16x16x32_bf16(false, a, false, b1, (short)0, acc1, false, false);
    acc2 = __builtin_amdgcn_wmma_f32_16x16x32_bf16(false, a, false, b2, (short)0, acc2, false, false);
    acc3 = __builtin_amdgcn_wmma_f32_16x16x32_bf16(false, a, false, b3, (short)0, acc3, false, false);
    ap += 1024; bp0 += 1024; bp2 += 1024;
  }
  const int idx = lane & 15, grp = lane >> 4;
  const int hid = htile * 16 + idx;
  const float bi = bias[hid], bf = bias[H_ + hid], bg = bias[2 * H_ + hid], bo = bias[3 * H_ + hid];
#pragma unroll
  for (int r = 0; r < 8; ++r) {
    int row = mt * 16 + grp * 8 + r;               // CDNA5 f32 C/D layout
    float ig = sigf(acc0[r] + bi);
    float fg = sigf(acc1[r] + bf);
    float gg = tanhf(acc2[r] + bg);
    float og = sigf(acc3[r] + bo);
    float cn = fg * cst[row * H_ + hid] + ig * gg;
    cst[row * H_ + hid] = cn;
    hst[row * H_ + hid] = og * tanhf(cn);
  }
}

// ---------- Gaussian-mixture attention (per batch row) ----------
__global__ __launch_bounds__(128) void attention(const float* __restrict__ h0,
                                                 const float* __restrict__ Watt,
                                                 const float* __restrict__ batt,
                                                 const float* __restrict__ ctxm,
                                                 float* __restrict__ wst,
                                                 float* __restrict__ kst,
                                                 float* __restrict__ o_phi,
                                                 float* __restrict__ o_alpha,
                                                 float* __restrict__ o_beta,
                                                 float* __restrict__ o_kappa,
                                                 int t) {
  __shared__ float abk[32];
  __shared__ float sphi[TC_];
  const int b = blockIdx.x;
  const int tid = threadIdx.x, lane = tid & 31, wave = tid >> 5;
  for (int o = wave; o < 3 * KA_; o += 4) {
    float s = 0.0f;
    for (int j = lane; j < H_; j += 32) s += h0[b * H_ + j] * Watt[o * H_ + j];
    for (int m = 16; m; m >>= 1) s += __shfl_xor(s, m, 32);
    if (lane == 0) abk[o] = __expf(s + batt[o]);
  }
  __syncthreads();
  if (tid < KA_) {
    float kap = abk[2 * KA_ + tid] + kst[b * KA_ + tid];
    abk[2 * KA_ + tid] = kap;
    kst[b * KA_ + tid] = kap;
    o_alpha[((size_t)b * TS_ + t) * KA_ + tid] = abk[tid];
    o_beta [((size_t)b * TS_ + t) * KA_ + tid] = abk[KA_ + tid];
    o_kappa[((size_t)b * TS_ + t) * KA_ + tid] = kap;
  }
  __syncthreads();
  {
    float u = (float)tid, p = 0.0f;
#pragma unroll
    for (int kk = 0; kk < KA_; ++kk) {
      float d = abk[2 * KA_ + kk] - u;
      p += abk[kk] * __expf(-d * d / abk[KA_ + kk]);
    }
    sphi[tid] = p;
    o_phi[((size_t)b * TS_ + t) * TC_ + tid] = p;
  }
  __syncthreads();
  {
    float acc = 0.0f;
    for (int u = 0; u < TC_; ++u) acc += sphi[u] * ctxm[((size_t)b * TC_ + u) * ENC_ + tid];
    wst[b * ENC_ + tid] = acc;
  }
}

// ---------- WMMA output projection ----------
__global__ __launch_bounds__(256) void outproj(const unsigned short* __restrict__ Apack,
                                               const unsigned short* __restrict__ Wpack,
                                               const float* __restrict__ bout,
                                               float* __restrict__ outs, int t) {
  const int wave = threadIdx.x >> 5, lane = threadIdx.x & 31;
  const int mt = blockIdx.x, nt = wave;
  v8f acc = {};
  const char* ap = (const char*)Apack + (((size_t)mt * KTO) * 32 + lane) * 32;
  const char* bp = (const char*)Wpack + (((size_t)nt * KTO) * 32 + lane) * 32;
#pragma unroll 2
  for (int kt = 0; kt < KTO; ++kt) {
    v16bf a = *(const v16bf*)ap;
    v16bf b = *(const v16bf*)bp;
    acc = __builtin_amdgcn_wmma_f32_16x16x32_bf16(false, a, false, b, (short)0, acc, false, false);
    ap += 1024; bp += 1024;
  }
  const int idx = lane & 15, grp = lane >> 4;
  const int n = nt * 16 + idx;
  if (n < NOUT_) {
    float bb = bout[n];
#pragma unroll
    for (int r = 0; r < 8; ++r) {
      int row = mt * 16 + grp * 8 + r;
      outs[((size_t)row * TS_ + t) * NOUT_ + n] = acc[r] + bb;
    }
  }
}

// ---------- final carry-state copy ----------
__global__ __launch_bounds__(256) void finalize(const float* __restrict__ hs,
                                                const float* __restrict__ cs,
                                                const float* __restrict__ wst,
                                                const float* __restrict__ kst,
                                                float* __restrict__ out) {
  int id = blockIdx.x * blockDim.x + threadIdx.x;
  const int NH = 3 * B_ * H_;
  const size_t OFF_HF = 18284544, OFF_CF = 18677760, OFF_WF = 19070976, OFF_KF = 19087360;
  if (id < NH)                       out[OFF_HF + id] = hs[id];
  else if (id < 2 * NH)              out[OFF_CF + (id - NH)] = cs[id - NH];
  else if (id < 2 * NH + B_ * ENC_)  out[OFF_WF + (id - 2 * NH)] = wst[id - 2 * NH];
  else if (id < 2 * NH + B_ * ENC_ + B_ * KA_)
                                     out[OFF_KF + (id - 2 * NH - B_ * ENC_)] = kst[id - 2 * NH - B_ * ENC_];
}

extern "C" void kernel_launch(void* const* d_in, const int* in_sizes, int n_in,
                              void* d_out, int out_size, void* d_ws, size_t ws_size,
                              hipStream_t stream) {
  (void)in_sizes; (void)n_in; (void)out_size; (void)ws_size;
  const float* strokes = (const float*)d_in[0];
  const float* context = (const float*)d_in[1];
  const float* cmask   = (const float*)d_in[2];
  const float* h0i     = (const float*)d_in[3];
  const float* c0i     = (const float*)d_in[4];
  const float* w0i     = (const float*)d_in[5];
  const float* k0i     = (const float*)d_in[6];
  const float* Wih0    = (const float*)d_in[7];
  const float* Whh0    = (const float*)d_in[8];
  const float* bih0    = (const float*)d_in[9];
  const float* bhh0    = (const float*)d_in[10];
  const float* Wihn    = (const float*)d_in[11];
  const float* Whhn    = (const float*)d_in[12];
  const float* bihn    = (const float*)d_in[13];
  const float* bhhn    = (const float*)d_in[14];
  const float* Watt    = (const float*)d_in[15];
  const float* batt    = (const float*)d_in[16];
  const float* Wout    = (const float*)d_in[17];
  const float* bout    = (const float*)d_in[18];
  float* out = (float*)d_out;

  // ---- workspace carve-up (f32 region, then aligned bf16 region) ----
  float* fws = (float*)d_ws;
  size_t off = 0;
  float* h_state = fws + off; off += (size_t)3 * B_ * H_;
  float* c_state = fws + off; off += (size_t)3 * B_ * H_;
  float* w_state = fws + off; off += (size_t)B_ * ENC_;
  float* k_state = fws + off; off += (size_t)B_ * KA_;
  float* bias0   = fws + off; off += G_;
  float* biasn   = fws + off; off += 2 * G_;
  float* ctxm    = fws + off; off += (size_t)B_ * TC_ * ENC_;
  off = (off + 63) & ~(size_t)63;          // 256B align for bf16 fragments
  unsigned short* hws = (unsigned short*)(fws + off);
  size_t ho = 0;
  unsigned short* W0p = hws + ho; ho += (size_t)256 * KT0 * 512;
  unsigned short* W1p = hws + ho; ho += (size_t)256 * KTN * 512;
  unsigned short* W2p = hws + ho; ho += (size_t)256 * KTN * 512;
  unsigned short* WOp = hws + ho; ho += (size_t)8 * KTO * 512;
  unsigned short* A0p = hws + ho; ho += (size_t)8 * KT0 * 512;
  unsigned short* A1p = hws + ho; ho += (size_t)8 * KTN * 512;
  unsigned short* A2p = hws + ho; ho += (size_t)8 * KTN * 512;
  unsigned short* AOp = hws + ho; ho += (size_t)8 * KTO * 512;

  float* o_outs  = out;
  float* o_phi   = out + 7929856;
  float* o_alpha = out + 16318464;
  float* o_beta  = out + 16973824;
  float* o_kappa = out + 17629184;

  // ---- one-time prep (deterministic; re-run every call) ----
  pack_w0  <<<18944, 256, 0, stream>>>(Wih0, Whh0, W0p);
  pack_wn  <<<35328, 256, 0, stream>>>(Wihn, Whhn, W1p);
  pack_wn  <<<35328, 256, 0, stream>>>(Wihn + (size_t)G_ * 1155, Whhn + (size_t)G_ * H_, W2p);
  pack_wout<<<1536,  256, 0, stream>>>(Wout, WOp);
  prep_bias<<<48,    256, 0, stream>>>(bih0, bhh0, bihn, bhhn, bias0, biasn);
  prep_ctx <<<8192,  256, 0, stream>>>(context, cmask, ctxm);
  init_state<<<3142, 256, 0, stream>>>(h0i, c0i, w0i, k0i, h_state, c_state, w_state, k_state);

  float* h1 = h_state + (size_t)B_ * H_;
  float* h2 = h_state + (size_t)2 * B_ * H_;
  float* c1 = c_state + (size_t)B_ * H_;
  float* c2 = c_state + (size_t)2 * B_ * H_;

  for (int t = 0; t < TS_; ++t) {
    assemble0 <<<592,  256, 0, stream>>>(strokes, w_state, h_state, A0p, t);
    lstm_gemm<KT0><<<64, 256, 0, stream>>>(A0p, W0p, bias0, h_state, c_state);
    attention <<<128,  128, 0, stream>>>(h_state, Watt, batt, ctxm, w_state, k_state,
                                         o_phi, o_alpha, o_beta, o_kappa, t);
    assemble12<<<1104, 256, 0, stream>>>(strokes, w_state, h_state, h1, A1p, t);
    lstm_gemm<KTN><<<64, 256, 0, stream>>>(A1p, W1p, biasn, h1, c1);
    assemble12<<<1104, 256, 0, stream>>>(strokes, w_state, h1, h2, A2p, t);
    lstm_gemm<KTN><<<64, 256, 0, stream>>>(A2p, W2p, biasn + G_, h2, c2);
    assembleOut<<<1536,256, 0, stream>>>(h_state, AOp);
    outproj   <<<8,    256, 0, stream>>>(AOp, WOp, bout, o_outs, t);
  }
  finalize<<<3142, 256, 0, stream>>>(h_state, c_state, w_state, k_state, out);
}
